// BitColumnParallelLinear_80187039416779
// MI455X (gfx1250) — compile-verified, matrix operand order checked by reading, not run on hardware
//
#include <hip/hip_runtime.h>
#include <cstdint>

typedef int v8i __attribute__((ext_vector_type(8)));

#define EPSQ 1e-5f
#define BM 128
#define BN 128
#define KTS 64
#define LDSTR 80   // padded LDS row stride (bytes); multiple of 16, 20-bank step

static __device__ __forceinline__ int clampi(int v, int lo, int hi) {
  return v < lo ? lo : (v > hi ? hi : v);
}

// ---------------- Kernel 1: partial sums of |w| ----------------
__global__ __launch_bounds__(256)
void wabs_partial_kernel(const float* __restrict__ w, float* __restrict__ partial,
                         long long n4) {
  __shared__ float sred[256];
  const float4* w4 = (const float4*)w;
  float s = 0.0f;
  long long stride = (long long)gridDim.x * blockDim.x;
  for (long long i = (long long)blockIdx.x * blockDim.x + threadIdx.x; i < n4; i += stride) {
    float4 v = w4[i];
    s += fabsf(v.x) + fabsf(v.y) + fabsf(v.z) + fabsf(v.w);
  }
  sred[threadIdx.x] = s;
  __syncthreads();
  for (int off = 128; off > 0; off >>= 1) {
    if ((int)threadIdx.x < off) sred[threadIdx.x] += sred[threadIdx.x + off];
    __syncthreads();
  }
  if (threadIdx.x == 0) partial[blockIdx.x] = sred[0];
}

// ---------------- Kernel 2: finalize weight scale ----------------
__global__ __launch_bounds__(256)
void wscale_finalize_kernel(const float* __restrict__ partial, float* __restrict__ scalars,
                            long long count) {
  __shared__ float sred[256];
  float s = 0.0f;
  for (int i = threadIdx.x; i < 1024; i += 256) s += partial[i];
  sred[threadIdx.x] = s;
  __syncthreads();
  for (int off = 128; off > 0; off >>= 1) {
    if ((int)threadIdx.x < off) sred[threadIdx.x] += sred[threadIdx.x + off];
    __syncthreads();
  }
  if (threadIdx.x == 0) {
    float mean = sred[0] / (float)count;
    float invW = fmaxf(mean, EPSQ);  // dequant factor = 1/scale
    scalars[0] = invW;
    scalars[1] = 1.0f / invW;        // quant scale
  }
}

// ---------------- Kernel 3: ternarize weights to int8 ----------------
__global__ __launch_bounds__(256)
void wquant_kernel(const float* __restrict__ w, int8_t* __restrict__ wq,
                   const float* __restrict__ scalars, long long n4) {
  const float s = scalars[1];
  const float4* w4 = (const float4*)w;
  int* out4 = (int*)wq;
  long long stride = (long long)gridDim.x * blockDim.x;
  for (long long i = (long long)blockIdx.x * blockDim.x + threadIdx.x; i < n4; i += stride) {
    float4 v = w4[i];
    int a = clampi((int)rintf(v.x * s), -1, 1);
    int b = clampi((int)rintf(v.y * s), -1, 1);
    int c = clampi((int)rintf(v.z * s), -1, 1);
    int d = clampi((int)rintf(v.w * s), -1, 1);
    out4[i] = (a & 255) | ((b & 255) << 8) | ((c & 255) << 16) | ((d & 255) << 24);
  }
}

// ---------------- Kernel 4: per-token int8 absmax quantization ----------------
__global__ __launch_bounds__(256)
void aquant_kernel(const float* __restrict__ x, int8_t* __restrict__ xq,
                   float* __restrict__ scaleX, int K) {
  __shared__ float sred[256];
  const long long row = blockIdx.x;
  const float4* xr = (const float4*)(x + row * (long long)K);
  const int k4 = K >> 2;
  float amax = 0.0f;
  for (int i = threadIdx.x; i < k4; i += 256) {
    float4 v = xr[i];
    amax = fmaxf(amax, fmaxf(fmaxf(fabsf(v.x), fabsf(v.y)), fmaxf(fabsf(v.z), fabsf(v.w))));
  }
  sred[threadIdx.x] = amax;
  __syncthreads();
  for (int off = 128; off > 0; off >>= 1) {
    if ((int)threadIdx.x < off)
      sred[threadIdx.x] = fmaxf(sred[threadIdx.x], sred[threadIdx.x + off]);
    __syncthreads();
  }
  const float am = fmaxf(sred[0], EPSQ);
  const float s = 127.0f / am;
  if (threadIdx.x == 0) scaleX[row] = am * (1.0f / 127.0f);  // dequant factor
  int* outr = (int*)(xq + row * (long long)K);
  for (int i = threadIdx.x; i < k4; i += 256) {
    float4 v = xr[i];
    int a = clampi((int)rintf(v.x * s), -128, 127);
    int b = clampi((int)rintf(v.y * s), -128, 127);
    int c = clampi((int)rintf(v.z * s), -128, 127);
    int d = clampi((int)rintf(v.w * s), -128, 127);
    outr[i] = (a & 255) | ((b & 255) << 8) | ((c & 255) << 16) | ((d & 255) << 24);
  }
}

// ---------------- Async copy helpers (CDNA5 GLOBAL_LOAD_ASYNC_TO_LDS) ----------------
// dst operand = VGPR with 32-bit LDS byte address; src = VGPR pair with 64-bit
// global address; tracked by ASYNCcnt (no loadcnt stall, no VGPR round-trip).
static __device__ __forceinline__ void async_copy_b128(void* lds_dst, const void* gsrc) {
  uint32_t l = (uint32_t)(uintptr_t)lds_dst;   // generic low 32 bits = LDS offset
  uint64_t g = (uint64_t)(uintptr_t)gsrc;
  asm volatile("global_load_async_to_lds_b128 %0, %1, off" : : "v"(l), "v"(g) : "memory");
}

// ---------------- Fragment load from LDS per ISA 8-bit A/B layout ----------------
// lane L: matrix row/col = L%16, half h = L/16; dword d covers K bytes at
// {0,4,16,20,32,36,48,52}[d] + 8*h  => four 8-byte loads at p+0,16,32,48 with p+=8*h.
static __device__ __forceinline__ v8i load_frag(const char* p) {
  int2 q0 = *(const int2*)(p + 0);
  int2 q1 = *(const int2*)(p + 16);
  int2 q2 = *(const int2*)(p + 32);
  int2 q3 = *(const int2*)(p + 48);
  v8i f;
  f[0] = q0.x; f[1] = q0.y; f[2] = q1.x; f[3] = q1.y;
  f[4] = q2.x; f[5] = q2.y; f[6] = q3.x; f[7] = q3.y;
  return f;
}

// ---------------- Kernel 5: int8 x int8 -> int32 WMMA GEMM ----------------
// out[m,n] = (sum_k xq[m,k]*wq[n,k]) * scaleX[m] * invW + bias[n]
__global__ __launch_bounds__(256, 1)
void bitgemm_kernel(const int8_t* __restrict__ Aq, const int8_t* __restrict__ Bq,
                    const float* __restrict__ scaleX, const float* __restrict__ scalars,
                    const float* __restrict__ bias, float* __restrict__ out,
                    int M, int N, int K) {
  __shared__ __align__(16) char smA[2][BM * LDSTR];
  __shared__ __align__(16) char smB[2][BN * LDSTR];

  const int tid  = threadIdx.x;
  const int lane = tid & 31;
  const int wid  = tid >> 5;
  const int wm   = wid & 1;    // 2 waves across M (64 rows each)
  const int wn   = wid >> 1;   // 4 waves across N (32 cols each)
  const int lm   = lane & 15;
  const int lh   = lane >> 4;

  const int nBase = blockIdx.x * BN;
  const int mBase = blockIdx.y * BM;

  // per-thread staging coordinates: 2 x 16B per matrix per K-step
  const int ib0  = tid << 4;
  const int row0 = ib0 >> 6, col0 = ib0 & 63;
  const int ib1  = (tid + 256) << 4;
  const int row1 = ib1 >> 6, col1 = ib1 & 63;

  const int8_t* gA0 = Aq + (size_t)(mBase + row0) * K + col0;
  const int8_t* gA1 = Aq + (size_t)(mBase + row1) * K + col1;
  const int8_t* gB0 = Bq + (size_t)(nBase + row0) * K + col0;
  const int8_t* gB1 = Bq + (size_t)(nBase + row1) * K + col1;
  char* lA0base[2] = {&smA[0][row0 * LDSTR + col0], &smA[1][row0 * LDSTR + col0]};
  char* lA1base[2] = {&smA[0][row1 * LDSTR + col1], &smA[1][row1 * LDSTR + col1]};
  char* lB0base[2] = {&smB[0][row0 * LDSTR + col0], &smB[1][row0 * LDSTR + col0]};
  char* lB1base[2] = {&smB[0][row1 * LDSTR + col1], &smB[1][row1 * LDSTR + col1]};

  v8i acc[4][2];
  const v8i vzero = {0, 0, 0, 0, 0, 0, 0, 0};
#pragma unroll
  for (int i = 0; i < 4; ++i)
#pragma unroll
    for (int j = 0; j < 2; ++j) acc[i][j] = vzero;

  auto preload = [&](int buf, int t) {
    const size_t kk = (size_t)t * KTS;
    async_copy_b128(lA0base[buf], gA0 + kk);
    async_copy_b128(lA1base[buf], gA1 + kk);
    async_copy_b128(lB0base[buf], gB0 + kk);
    async_copy_b128(lB1base[buf], gB1 + kk);
  };

  auto compute_tile = [&](int buf) {
    v8i afr[4], bfr[2];
#pragma unroll
    for (int mt = 0; mt < 4; ++mt)
      afr[mt] = load_frag(&smA[buf][(wm * 64 + mt * 16 + lm) * LDSTR + lh * 8]);
#pragma unroll
    for (int nt = 0; nt < 2; ++nt)
      bfr[nt] = load_frag(&smB[buf][(wn * 32 + nt * 16 + lm) * LDSTR + lh * 8]);
#pragma unroll
    for (int mt = 0; mt < 4; ++mt)
#pragma unroll
      for (int nt = 0; nt < 2; ++nt)
        acc[mt][nt] = __builtin_amdgcn_wmma_i32_16x16x64_iu8(
            /*sgn_a=*/true, afr[mt], /*sgn_b=*/true, bfr[nt], acc[mt][nt],
            /*reuse_a=*/false, /*reuse_b=*/false);
  };

  const int nT = K / KTS;
  preload(0, 0);
  for (int t = 0; t < nT - 1; ++t) {
    const int buf = t & 1;
    preload(buf ^ 1, t + 1);                          // 4 newer async ops in flight
    asm volatile("s_wait_asynccnt 4" ::: "memory");   // current buffer's 4 ops done
    __syncthreads();                                  // all waves' copies visible
    compute_tile(buf);
    __syncthreads();                                  // done reading buf before reuse
  }
  asm volatile("s_wait_asynccnt 0" ::: "memory");
  __syncthreads();
  compute_tile((nT - 1) & 1);

  // epilogue: dequantize + bias. C/D layout: VGPR r -> row r + 8*h, lane%16 -> col
  const float invW = scalars[0];
#pragma unroll
  for (int mt = 0; mt < 4; ++mt) {
    const size_t m0 = (size_t)mBase + wm * 64 + mt * 16 + lh * 8;
#pragma unroll
    for (int nt = 0; nt < 2; ++nt) {
      const int n = nBase + wn * 32 + nt * 16 + lm;
      const float bn = bias[n];
      const v8i a = acc[mt][nt];
#pragma unroll
      for (int r = 0; r < 8; ++r) {
        const size_t m = m0 + r;
        out[m * (size_t)N + n] = (float)a[r] * (scaleX[m] * invW) + bn;
      }
    }
  }
}

extern "C" void kernel_launch(void* const* d_in, const int* in_sizes, int n_in,
                              void* d_out, int out_size, void* d_ws, size_t ws_size,
                              hipStream_t stream) {
  const float* x    = (const float*)d_in[0];
  const float* w    = (const float*)d_in[1];
  const float* bias = (const float*)d_in[2];
  float* out = (float*)d_out;

  const int K = 4096;
  const int N = in_sizes[2];        // 16384
  const int M = in_sizes[0] / K;    // 8192 (= B*S)

  char* ws = (char*)d_ws;
  int8_t* xq      = (int8_t*)ws;                                   // M*K bytes
  int8_t* wq      = (int8_t*)(ws + (size_t)M * K);                 // N*K bytes
  float*  scaleX  = (float*)(ws + (size_t)M * K + (size_t)N * K);  // M floats
  float*  partial = scaleX + M;                                    // 1024 floats
  float*  scalars = partial + 1024;                                // 2 floats

  const long long wcount = (long long)N * K;

  wabs_partial_kernel<<<1024, 256, 0, stream>>>(w, partial, wcount / 4);
  wscale_finalize_kernel<<<1, 256, 0, stream>>>(partial, scalars, wcount);
  wquant_kernel<<<2048, 256, 0, stream>>>(w, wq, scalars, wcount / 4);
  aquant_kernel<<<M, 256, 0, stream>>>(x, xq, scaleX, K);

  dim3 grid(N / BN, M / BM);
  bitgemm_kernel<<<grid, 256, 0, stream>>>(xq, wq, scaleX, scalars, bias, out, M, N, K);
}
